// CrossAttention_54348516163802
// MI455X (gfx1250) — matrix operand order repaired and checked by reading, not compiled, not run on hardware
//
#include <hip/hip_runtime.h>
#include <hip/hip_bf16.h>
#include <stdint.h>

// ---------------------------------------------------------------------------
// CDNA5 (gfx1250) cross-attention: f16 WMMA everywhere, flash-attention core,
// async global->LDS staging for K tiles. Wave32. WMMA 16x16x32 f16 -> f32.
// ---------------------------------------------------------------------------

typedef __attribute__((ext_vector_type(16))) _Float16 v16h;
typedef __attribute__((ext_vector_type(8)))  float    v8f;

union Frag {
  v16h v;
  _Float16 h[16];
  uint4 q[2];
};

__device__ __forceinline__ v8f wmma16(v16h a, v16h b, v8f c) {
  return __builtin_amdgcn_wmma_f32_16x16x32_f16(
      /*neg_a=*/false, a, /*neg_b=*/false, b,
      /*c_mod=*/(short)0, c, /*reuse_a=*/false, /*reuse_b=*/false);
}

constexpr int Bb = 4, Nn = 2048, Mm = 2048, Cc = 1024, Hh = 16, Dd = 64;
constexpr float SCALE = 0.125f;  // D^-0.5 = 64^-0.5

// ---------------------------------------------------------------------------
// f32 -> f16 convert, 4 elements/thread (float4 in, 8B out)
// ---------------------------------------------------------------------------
__global__ void ca_cvt_f16(const float* __restrict__ in,
                           _Float16* __restrict__ out, int n4) {
  int i = blockIdx.x * blockDim.x + threadIdx.x;
  int stride = gridDim.x * blockDim.x;
  for (; i < n4; i += stride) {
    float4 f = ((const float4*)in)[i];
    union { _Float16 h[4]; uint2 u; } t;
    t.h[0] = (_Float16)f.x; t.h[1] = (_Float16)f.y;
    t.h[2] = (_Float16)f.z; t.h[3] = (_Float16)f.w;
    *(uint2*)(out + 4 * (size_t)i) = t.u;
  }
}

// ---------------------------------------------------------------------------
// WMMA GEMM: out[r,c] = sum_k A[r,k]*W[k,c] + bias[c]
// A: [rows,K] f16.  W: [K,ldw] f16.  Output ld = ldw.
// Block: 256 thr (8 waves), tile 256 rows x 64 cols; wave tile 32x64
// (8 WMMAs / K-step). K step 32.
// ---------------------------------------------------------------------------
template <bool F16OUT>
__global__ __launch_bounds__(256) void ca_gemm_wmma(
    const _Float16* __restrict__ A, const _Float16* __restrict__ W,
    const float* __restrict__ bias, void* __restrict__ outp, int K, int ldw) {
  __shared__ __align__(16) _Float16 Al[256 * 40];  // [row][k], pad 40
  __shared__ __align__(16) _Float16 Wl[64 * 40];   // [c][k] transposed, pad 40

  const int tid = threadIdx.x;
  const int wave = tid >> 5, lane = tid & 31;
  const int l16 = lane & 15, half = lane >> 4;
  const int c0 = blockIdx.x * 64;
  const size_t r0 = (size_t)blockIdx.y * 256;

  v8f acc[2][4];
#pragma unroll
  for (int rh = 0; rh < 2; ++rh)
    for (int ds = 0; ds < 4; ++ds)
      for (int i = 0; i < 8; ++i) acc[rh][ds][i] = 0.f;

  const int wc = tid & 63, wkg = tid >> 6;  // W staging coords

  for (int kt = 0; kt < K; kt += 32) {
    // stage A tile 256x32: one row (4 x uint4 = one 64B line) per thread
    const uint4* asrc = (const uint4*)(A + (r0 + tid) * (size_t)K + kt);
#pragma unroll
    for (int j = 0; j < 4; ++j) *(uint4*)&Al[tid * 40 + 8 * j] = asrc[j];
    // stage W tile 32x64 transposed into [c][k] (coalesced across 64 lanes)
#pragma unroll
    for (int j = 0; j < 8; ++j) {
      int kl = wkg * 8 + j;
      Wl[wc * 40 + kl] = W[(size_t)(kt + kl) * ldw + c0 + wc];
    }
    __syncthreads();

    // preload all fragments, then issue the 8-WMMA group
    Frag aF[2];
#pragma unroll
    for (int rh = 0; rh < 2; ++rh) {
      const _Float16* ap = &Al[(wave * 32 + rh * 16 + l16) * 40];
      aF[rh].q[0] = *(const uint4*)(ap + 16 * half);
      aF[rh].q[1] = *(const uint4*)(ap + 16 * half + 8);
    }
    Frag wF[4];
#pragma unroll
    for (int ds = 0; ds < 4; ++ds) {
      const _Float16* wp = &Wl[(ds * 16 + l16) * 40 + 16 * half];
      wF[ds].q[0] = *(const uint4*)(wp);
      wF[ds].q[1] = *(const uint4*)(wp + 8);
    }
#pragma unroll
    for (int ds = 0; ds < 4; ++ds)
#pragma unroll
      for (int rh = 0; rh < 2; ++rh)
        acc[rh][ds] = wmma16(aF[rh].v, wF[ds].v, acc[rh][ds]);
    __syncthreads();
  }

  // epilogue: C/D layout row = i + 8*half, col = lane%16
#pragma unroll
  for (int rh = 0; rh < 2; ++rh)
#pragma unroll
    for (int ds = 0; ds < 4; ++ds) {
      int c = c0 + ds * 16 + l16;
      float bv = bias[c];
#pragma unroll
      for (int i = 0; i < 8; ++i) {
        size_t r = r0 + wave * 32 + rh * 16 + i + 8 * half;
        float val = acc[rh][ds][i] + bv;
        size_t idx = r * (size_t)ldw + c;
        if constexpr (F16OUT) ((_Float16*)outp)[idx] = (_Float16)val;
        else                  ((float*)outp)[idx] = val;
      }
    }
}

// ---------------------------------------------------------------------------
// Flash attention: workgroup = 128 Q rows of one (b,h); online softmax.
// K chunk staged via GLOBAL_LOAD_ASYNC_TO_LDS_B128 (ASYNCcnt), V staged
// transposed manually (async copy cannot reorder).
// ---------------------------------------------------------------------------
__global__ __launch_bounds__(256) void ca_flash_attn(
    const _Float16* __restrict__ Qh, const _Float16* __restrict__ KVh,
    const float* __restrict__ amask, const unsigned char* __restrict__ kpm,
    _Float16* __restrict__ Oh) {
  __shared__ __align__(16) _Float16 Kl[32 * 72];     // [m][d], pad 72
  __shared__ __align__(16) _Float16 Vl[64 * 40];     // [d][m] transposed
  __shared__ __align__(16) _Float16 Pl[8][16 * 40];  // per-wave P scratch

  const int tid = threadIdx.x;
  const int wave = tid >> 5, lane = tid & 31;
  const int l16 = lane & 15, half = lane >> 4;
  const int bh = blockIdx.y;
  const int b = bh >> 4;  // Hh = 16
  const int h = bh & 15;
  const int nw = blockIdx.x * 128 + wave * 16;

  // Q A-fragments, held in registers for the whole M loop
  const _Float16* qrow = Qh + ((size_t)(b * Nn + nw + l16)) * Cc + h * Dd;
  Frag qa[2];
#pragma unroll
  for (int dc = 0; dc < 2; ++dc) {
    qa[dc].q[0] = *(const uint4*)(qrow + dc * 32 + 16 * half);
    qa[dc].q[1] = *(const uint4*)(qrow + dc * 32 + 16 * half + 8);
  }

  float mrow[8], lrow[8];
  v8f o[4];
#pragma unroll
  for (int i = 0; i < 8; ++i) { mrow[i] = -3.0e38f; lrow[i] = 0.f; }
  for (int d = 0; d < 4; ++d)
    for (int i = 0; i < 8; ++i) o[d][i] = 0.f;

  const _Float16* Kg = KVh + (size_t)(b * Mm) * (2 * Cc) + h * Dd;
  const _Float16* Vg = Kg + Cc;
  const unsigned char* kb = kpm + b * Mm;

  const int sml = tid >> 3;       // staging m-row 0..31
  const int sd0 = (tid & 7) * 8;  // staging d-col 0,8,..,56
  // per-thread LDS byte address for async K staging (generic addr low 32 bits
  // are the wave-relative LDS offset, per ISA aperture truncation)
  const unsigned kdst = (unsigned)(uintptr_t)(&Kl[sml * 72 + sd0]);

  for (int mc = 0; mc < Mm; mc += 32) {
    // stage K chunk [m][d]: async DMA global -> LDS, 16B per lane
    {
      unsigned long long ksrc =
          (unsigned long long)(Kg + (size_t)(mc + sml) * (2 * Cc) + sd0);
      asm volatile("global_load_async_to_lds_b128 %0, %1, off"
                   :: "v"(kdst), "v"(ksrc)
                   : "memory");
    }
    // stage V chunk transposed to [d][m] (manual: reorders bytes)
    {
      union { uint4 q; _Float16 e[8]; } t;
      t.q = *(const uint4*)(Vg + (size_t)(mc + sml) * (2 * Cc) + sd0);
#pragma unroll
      for (int j = 0; j < 8; ++j) Vl[(sd0 + j) * 40 + sml] = t.e[j];
    }
    asm volatile("s_wait_asynccnt 0x0" ::: "memory");
    __syncthreads();

    // S(16x32) = Q(16x64) @ K^T : preload 4 B-frags, then 4 WMMAs
    Frag kf[2][2];
#pragma unroll
    for (int ms = 0; ms < 2; ++ms)
#pragma unroll
      for (int dc = 0; dc < 2; ++dc) {
        const _Float16* kp = &Kl[(ms * 16 + l16) * 72 + dc * 32 + 16 * half];
        kf[ms][dc].q[0] = *(const uint4*)(kp);
        kf[ms][dc].q[1] = *(const uint4*)(kp + 8);
      }
    v8f s[2];
    for (int ms = 0; ms < 2; ++ms)
      for (int i = 0; i < 8; ++i) s[ms][i] = 0.f;
#pragma unroll
    for (int ms = 0; ms < 2; ++ms)
#pragma unroll
      for (int dc = 0; dc < 2; ++dc)
        s[ms] = wmma16(qa[dc].v, kf[ms][dc].v, s[ms]);

    // scale + attn_mask + key_padding_mask
    float sv[2][8];
    bool mskd[2];
#pragma unroll
    for (int ms = 0; ms < 2; ++ms) {
      int col = mc + ms * 16 + l16;
      mskd[ms] = kb[col] != 0;
      const float* am = amask + (size_t)(nw + 8 * half) * Mm + col;
#pragma unroll
      for (int i = 0; i < 8; ++i) {
        float v = s[ms][i] * SCALE + am[(size_t)i * Mm];
        sv[ms][i] = mskd[ms] ? -3.0e38f : v;
      }
    }

    // online softmax (row stats replicated across each 16-lane group)
    float pv[2][8];
#pragma unroll
    for (int i = 0; i < 8; ++i) {
      float cm = fmaxf(sv[0][i], sv[1][i]);
#pragma unroll
      for (int off = 1; off < 16; off <<= 1)
        cm = fmaxf(cm, __shfl_xor(cm, off, 16));
      float mnew = fmaxf(mrow[i], cm);
      float corr = __expf(mrow[i] - mnew);
      float p0 = mskd[0] ? 0.f : __expf(sv[0][i] - mnew);
      float p1 = mskd[1] ? 0.f : __expf(sv[1][i] - mnew);
      pv[0][i] = p0; pv[1][i] = p1;
      float cs = p0 + p1;
#pragma unroll
      for (int off = 1; off < 16; off <<= 1)
        cs += __shfl_xor(cs, off, 16);
      lrow[i] = lrow[i] * corr + cs;
      mrow[i] = mnew;
#pragma unroll
      for (int d = 0; d < 4; ++d) o[d][i] *= corr;
    }

    // relayout P (C-layout) -> A-fragment via per-wave LDS scratch
#pragma unroll
    for (int i = 0; i < 8; ++i) {
      _Float16* pr = &Pl[wave][(i + 8 * half) * 40 + l16];
      pr[0]  = (_Float16)pv[0][i];
      pr[16] = (_Float16)pv[1][i];
    }
    asm volatile("s_wait_dscnt 0x0" ::: "memory");  // wave-local LDS RAW

    Frag pf;
    const _Float16* pp = &Pl[wave][l16 * 40];
    pf.q[0] = *(const uint4*)(pp + 16 * half);
    pf.q[1] = *(const uint4*)(pp + 16 * half + 8);

    // O(16x64) += P(16x32) @ V(32x64): preload 4 B-frags, then 4 WMMAs
    Frag vf[4];
#pragma unroll
    for (int ds = 0; ds < 4; ++ds) {
      const _Float16* vp = &Vl[(ds * 16 + l16) * 40 + 16 * half];
      vf[ds].q[0] = *(const uint4*)(vp);
      vf[ds].q[1] = *(const uint4*)(vp + 8);
    }
#pragma unroll
    for (int ds = 0; ds < 4; ++ds) o[ds] = wmma16(pf.v, vf[ds].v, o[ds]);

    __syncthreads();  // Kl/Vl reused next chunk
  }

  // normalize and write O (f16, [B*N, C])
#pragma unroll
  for (int ds = 0; ds < 4; ++ds)
#pragma unroll
    for (int i = 0; i < 8; ++i) {
      float val = o[ds][i] / lrow[i];
      Oh[(size_t)(b * Nn + nw + i + 8 * half) * Cc + h * Dd + ds * 16 + l16] =
          (_Float16)val;
    }
}

// ---------------------------------------------------------------------------
// Workspace layout (bytes; total ~104 MB)
// ---------------------------------------------------------------------------
constexpr size_t OFF_XH  = 0;
constexpr size_t OFF_CH  = OFF_XH  + (size_t)2 * Bb * Nn * Cc;
constexpr size_t OFF_WQ  = OFF_CH  + (size_t)2 * Bb * Mm * Cc;
constexpr size_t OFF_WKV = OFF_WQ  + (size_t)2 * Cc * Cc;
constexpr size_t OFF_WP  = OFF_WKV + (size_t)2 * Cc * 2 * Cc;
constexpr size_t OFF_QH  = OFF_WP  + (size_t)2 * Cc * Cc;
constexpr size_t OFF_KVH = OFF_QH  + (size_t)2 * Bb * Nn * Cc;
constexpr size_t OFF_OH  = OFF_KVH + (size_t)2 * Bb * Mm * 2 * Cc;

extern "C" void kernel_launch(void* const* d_in, const int* in_sizes, int n_in,
                              void* d_out, int out_size, void* d_ws,
                              size_t ws_size, hipStream_t stream) {
  const float* x      = (const float*)d_in[0];
  const float* ctx    = (const float*)d_in[1];
  const unsigned char* kpm = (const unsigned char*)d_in[2];  // jax bool: 1 B
  const float* amask  = (const float*)d_in[3];
  const float* Wq     = (const float*)d_in[4];
  const float* bq     = (const float*)d_in[5];
  const float* Wkv    = (const float*)d_in[6];
  const float* bkv    = (const float*)d_in[7];
  const float* Wproj  = (const float*)d_in[8];
  const float* bproj  = (const float*)d_in[9];
  float* out = (float*)d_out;

  char* ws = (char*)d_ws;
  _Float16* xh   = (_Float16*)(ws + OFF_XH);
  _Float16* ch   = (_Float16*)(ws + OFF_CH);
  _Float16* wqh  = (_Float16*)(ws + OFF_WQ);
  _Float16* wkvh = (_Float16*)(ws + OFF_WKV);
  _Float16* wph  = (_Float16*)(ws + OFF_WP);
  _Float16* qh   = (_Float16*)(ws + OFF_QH);
  _Float16* kvh  = (_Float16*)(ws + OFF_KVH);
  _Float16* oh   = (_Float16*)(ws + OFF_OH);

  // 1) convert inputs/weights to f16 (4 elems/thread)
  ca_cvt_f16<<<dim3(2048), 256, 0, stream>>>(x,     xh,   Bb * Nn * Cc / 4);
  ca_cvt_f16<<<dim3(2048), 256, 0, stream>>>(ctx,   ch,   Bb * Mm * Cc / 4);
  ca_cvt_f16<<<dim3(512),  256, 0, stream>>>(Wq,    wqh,  Cc * Cc / 4);
  ca_cvt_f16<<<dim3(1024), 256, 0, stream>>>(Wkv,   wkvh, Cc * 2 * Cc / 4);
  ca_cvt_f16<<<dim3(512),  256, 0, stream>>>(Wproj, wph,  Cc * Cc / 4);

  // 2) Q = x @ Wq + bq (f16 out)
  ca_gemm_wmma<true><<<dim3(16, 32), 256, 0, stream>>>(xh, wqh, bq, qh,
                                                       Cc, Cc);
  // 3) KV = ctx @ Wkv + bkv (f16 out; K = cols 0..1023, V = cols 1024..2047)
  ca_gemm_wmma<true><<<dim3(32, 32), 256, 0, stream>>>(ch, wkvh, bkv, kvh,
                                                       Cc, 2 * Cc);
  // 4) flash attention -> O (f16)
  ca_flash_attn<<<dim3(Nn / 128, Bb * Hh), 256, 0, stream>>>(qh, kvh, amask,
                                                             kpm, oh);
  // 5) out = O @ Wproj + bproj (f32 out)
  ca_gemm_wmma<false><<<dim3(16, 32), 256, 0, stream>>>(oh, wph, bproj, out,
                                                        Cc, Cc);
}
